// MultiHeadAttention_2362232013108
// MI455X (gfx1250) — matrix operand order repaired and checked by reading, not compiled, not run on hardware
//
#include <hip/hip_runtime.h>
#include <hip/hip_bf16.h>
#include <cstddef>
#include <cstdint>

typedef __attribute__((ext_vector_type(16))) __bf16 v16bf;
typedef __attribute__((ext_vector_type(8)))  __bf16 v8bf;
typedef __attribute__((ext_vector_type(4)))  __bf16 v4bf;
typedef __attribute__((ext_vector_type(8)))  float  v8f;

#define GK 1024   // inner dim (EMBED)
#define GN 1024   // output dim
#define GM 32768  // rows (N*L)

// ---- CDNA5 async global->LDS (ASYNCcnt-tracked), per-lane 16B ----
__device__ __forceinline__ void async_copy_b128(const __bf16* g, __bf16* lds) {
  asm volatile("global_load_async_to_lds_b128 %0, %1, off"
               :: "v"((uint32_t)(uintptr_t)lds), "v"((uint64_t)(uintptr_t)g)
               : "memory");
}
__device__ __forceinline__ void async_wait_le4() {
  asm volatile("s_wait_asynccnt 0x4" ::: "memory");
}
__device__ __forceinline__ void async_wait_le0() {
  asm volatile("s_wait_asynccnt 0x0" ::: "memory");
}

// stage one 128x32 A tile + 32x128 B tile: 4 async issues per thread
__device__ __forceinline__ void stage4(const __bf16* gA, const __bf16* gB,
                                       __bf16* lA, __bf16* lB) {
  async_copy_b128(gA,           lA);             // A rows 0..63
  async_copy_b128(gA + 64 * GK, lA + 64 * 32);   // A rows 64..127
  async_copy_b128(gB,           lB);             // B k-rows 0..15
  async_copy_b128(gB + 16 * GN, lB + 16 * 128);  // B k-rows 16..31
}

// 8 WMMAs on one staged tile pair
__device__ __forceinline__ void mma_step(const __bf16* sA, const __bf16* sB,
                                         v8f (&acc)[4][2], int lane, int wm, int wn) {
  v16bf af[4], bfr[2];
  const int mr = wm + (lane & 15);
  const int kb = (lane >> 4) * 8;   // A frag: lanes 0-15 K{0-7,16-23}; 16-31 K{8-15,24-31}
#pragma unroll
  for (int i = 0; i < 4; ++i) {
    union { v16bf v; v8bf h[2]; } u;
    u.h[0] = *(const v8bf*)&sA[(mr + i * 16) * 32 + kb];
    u.h[1] = *(const v8bf*)&sA[(mr + i * 16) * 32 + kb + 16];
    af[i] = u.v;
  }
#pragma unroll
  for (int j = 0; j < 2; ++j)
#pragma unroll
    for (int s = 0; s < 16; ++s)
      bfr[j][s] = sB[(kb + (s & 7) + (s >> 3) * 16) * 128 + wn + j * 16 + (lane & 15)];  // lane&15 = N col, K as A frag
#pragma unroll
  for (int i = 0; i < 4; ++i)
#pragma unroll
    for (int j = 0; j < 2; ++j)
      acc[i][j] = __builtin_amdgcn_wmma_f32_16x16x32_bf16(
          false, af[i], false, bfr[j], (short)0, acc[i][j], false, false);
}

// -------- prepass: f32 -> bf16 --------
__global__ __launch_bounds__(256) void f32_to_bf16(const float* __restrict__ in,
                                                   __bf16* __restrict__ out) {
  const size_t i = ((size_t)blockIdx.x * 256 + threadIdx.x) * 8;
  const float4 a = *(const float4*)(in + i);
  const float4 b = *(const float4*)(in + i + 4);
  v8bf o;
  o[0] = (__bf16)a.x; o[1] = (__bf16)a.y; o[2] = (__bf16)a.z; o[3] = (__bf16)a.w;
  o[4] = (__bf16)b.x; o[5] = (__bf16)b.y; o[6] = (__bf16)b.z; o[7] = (__bf16)b.w;
  *(v8bf*)(out + i) = o;
}

// -------- prepass: W[N][K] f32 -> Wt[K][N] bf16 --------
__global__ __launch_bounds__(256) void transpose_w_bf16(const float* __restrict__ W,
                                                        __bf16* __restrict__ Wt) {
  __shared__ float tile[32][33];
  const int bk = blockIdx.x * 32;
  const int bn = blockIdx.y * 32;
  const int tx = threadIdx.x & 31;
  const int ty = threadIdx.x >> 5;
#pragma unroll
  for (int r = 0; r < 4; ++r)
    tile[ty + r * 8][tx] = W[(size_t)(bn + ty + r * 8) * GK + bk + tx];
  __syncthreads();
#pragma unroll
  for (int r = 0; r < 4; ++r)
    Wt[(size_t)(bk + ty + r * 8) * GN + bn + tx] = (__bf16)tile[tx][ty + r * 8];
}

// -------- main GEMM: C[M,N] = A[M,K](bf16) * Bt[K,N](bf16), double-buffered --------
template <bool BIAS, bool OUT_BF16>
__global__ __launch_bounds__(256) void gemm_bf16_wmma(
    const __bf16* __restrict__ A, const __bf16* __restrict__ Bt,
    const float* __restrict__ bias, void* __restrict__ Cout) {
  __shared__ __attribute__((aligned(16))) __bf16 sA0[128 * 32];
  __shared__ __attribute__((aligned(16))) __bf16 sA1[128 * 32];
  __shared__ __attribute__((aligned(16))) __bf16 sB0[32 * 128];
  __shared__ __attribute__((aligned(16))) __bf16 sB1[32 * 128];

  const int tid  = threadIdx.x;
  const int lane = tid & 31;
  const int wave = tid >> 5;
  const int bm = blockIdx.y * 128;
  const int bn = blockIdx.x * 128;
  const int wm = (wave & 1) * 64;
  const int wn = (wave >> 1) * 32;

  v8f acc[4][2] = {};

  // per-thread staging coordinates (2 chunks of 16B per tile per array)
  const __bf16* gA = A + (size_t)(bm + (tid >> 2)) * GK + (tid & 3) * 8;
  const __bf16* gB = Bt + (size_t)(tid >> 4) * GN + bn + (tid & 15) * 8;
  const int aoff = (tid >> 2) * 32 + (tid & 3) * 8;
  const int boff = (tid >> 4) * 128 + (tid & 15) * 8;
  __bf16* la0 = &sA0[aoff];
  __bf16* la1 = &sA1[aoff];
  __bf16* lb0 = &sB0[boff];
  __bf16* lb1 = &sB1[boff];

  stage4(gA, gB, la0, lb0);                 // tile 0 -> buf0
  gA += 32; gB += (size_t)32 * GN;

  for (int it = 0; it < 32; it += 2) {
    // stage tile it+1 -> buf1; compute tile it from buf0
    stage4(gA, gB, la1, lb1);
    gA += 32; gB += (size_t)32 * GN;
    async_wait_le4();
    __syncthreads();
    mma_step(sA0, sB0, acc, lane, wm, wn);
    __syncthreads();

    // stage tile it+2 -> buf0; compute tile it+1 from buf1
    if (it + 2 < 32) {
      stage4(gA, gB, la0, lb0);
      gA += 32; gB += (size_t)32 * GN;
      async_wait_le4();
    } else {
      async_wait_le0();
    }
    __syncthreads();
    mma_step(sA1, sB1, acc, lane, wm, wn);
    __syncthreads();
  }

  // ---- epilogue: C/D layout VGPR r -> M = r + 8*(lane>=16), N = lane&15 ----
#pragma unroll
  for (int j = 0; j < 2; ++j) {
    const int col = bn + wn + j * 16 + (lane & 15);
    const float badd = BIAS ? bias[col] : 0.0f;
#pragma unroll
    for (int i = 0; i < 4; ++i) {
      const int row0 = bm + wm + i * 16 + (lane >> 4) * 8;
#pragma unroll
      for (int r = 0; r < 8; ++r) {
        const size_t idx = (size_t)(row0 + r) * GN + col;
        const float val = acc[i][j][r] + badd;
        if (OUT_BF16) ((__bf16*)Cout)[idx] = (__bf16)val;
        else          ((float*)Cout)[idx]  = val;
      }
    }
  }
}

// -------- attention: per (n,l) 16x16 Gram + softmax rowsum * head-sum(v) -----
__global__ __launch_bounds__(128) void attn_headmix(
    const __bf16* __restrict__ q, const __bf16* __restrict__ k,
    const __bf16* __restrict__ v, __bf16* __restrict__ o2) {
  __shared__ float shQ[4][1024];
  __shared__ float shK[4][1024];
  __shared__ float shS[4][16];

  const int lane = threadIdx.x & 31;
  const int wv   = threadIdx.x >> 5;
  const int pos  = blockIdx.x * 4 + wv;       // 0 .. 32767
  const int n = pos >> 12;
  const int l = pos & 4095;
  const size_t base = ((size_t)n * 4096 + l) * 1024;

  // cooperative load (bf16 -> f32) of Q,K rows into LDS
  const v4bf* qg = (const v4bf*)(q + base);
  const v4bf* kg = (const v4bf*)(k + base);
  float4* sq4 = (float4*)shQ[wv];
  float4* sk4 = (float4*)shK[wv];
#pragma unroll
  for (int t = 0; t < 8; ++t) {
    const v4bf a = qg[t * 32 + lane];
    const v4bf b = kg[t * 32 + lane];
    sq4[t * 32 + lane] = make_float4((float)a[0], (float)a[1], (float)a[2], (float)a[3]);
    sk4[t * 32 + lane] = make_float4((float)b[0], (float)b[1], (float)b[2], (float)b[3]);
  }
  __syncthreads();

  float e[8];
#pragma unroll
  for (int t = 0; t < 8; ++t) {
    const int a = t * 2 + (lane >> 4);
    const int b = lane & 15;
    const float* qr = &shQ[wv][a * 64];
    const float* kr = &shK[wv][b * 64];
    float d = 0.0f;
#pragma unroll 8
    for (int dd = 0; dd < 64; ++dd) d = fmaf(qr[dd], kr[dd], d);
    e[t] = d * (1.0f / 32.0f);
  }

  float vs0 = 0.0f, vs1 = 0.0f;
  const __bf16* vp = v + base;
#pragma unroll
  for (int b = 0; b < 16; ++b) {
    vs0 += (float)vp[b * 64 + lane];
    vs1 += (float)vp[b * 64 + lane + 32];
  }

#pragma unroll
  for (int t = 0; t < 8; ++t) {
    float m = e[t];
    m = fmaxf(m, __shfl_xor(m, 1));
    m = fmaxf(m, __shfl_xor(m, 2));
    m = fmaxf(m, __shfl_xor(m, 4));
    m = fmaxf(m, __shfl_xor(m, 8));
    const float ex = __expf(e[t] - m);
    float den = ex;
    den += __shfl_xor(den, 1);
    den += __shfl_xor(den, 2);
    den += __shfl_xor(den, 4);
    den += __shfl_xor(den, 8);
    float rs = ex / den;
    rs += __shfl_xor(rs, 1);
    rs += __shfl_xor(rs, 2);
    rs += __shfl_xor(rs, 4);
    rs += __shfl_xor(rs, 8);
    if ((lane & 15) == 0) shS[wv][t * 2 + (lane >> 4)] = rs;
  }
  __syncthreads();

  // out2[n, a*256 + l/16, (l%16)*64 + d] = s[a] * vsum[d]
  __bf16* op = o2 + (size_t)n * 4096 * 1024;
#pragma unroll
  for (int a = 0; a < 16; ++a) {
    const float s = shS[wv][a];
    const size_t ob = (size_t)(a * 256 + (l >> 4)) * 1024 + (size_t)(l & 15) * 64;
    op[ob + lane]      = (__bf16)(s * vs0);
    op[ob + lane + 32] = (__bf16)(s * vs1);
  }
}

extern "C" void kernel_launch(void* const* d_in, const int* in_sizes, int n_in,
                              void* d_out, int out_size, void* d_ws, size_t ws_size,
                              hipStream_t stream) {
  const float* x  = (const float*)d_in[0];
  const float* Wq = (const float*)d_in[1];
  const float* Wk = (const float*)d_in[2];
  const float* Wv = (const float*)d_in[3];
  const float* Wo = (const float*)d_in[4];
  const float* bo = (const float*)d_in[5];

  const size_t SZ = (size_t)GM * GK;   // activation tensor element count
  const size_t WZ = (size_t)GN * GK;   // weight element count

  __bf16* xb  = (__bf16*)d_ws;         // 64 MB
  __bf16* wtq = xb + SZ;               // 2 MB each
  __bf16* wtk = wtq + WZ;
  __bf16* wtv = wtk + WZ;
  __bf16* wto = wtv + WZ;
  __bf16* o2b = wto + WZ;              // 64 MB
  __bf16* qb  = o2b + SZ;              // 64 MB each
  __bf16* kb  = qb + SZ;
  __bf16* vb  = kb + SZ;

  f32_to_bf16<<<SZ / (256 * 8), 256, 0, stream>>>(x, xb);
  dim3 tgrid(GK / 32, GN / 32);
  transpose_w_bf16<<<tgrid, 256, 0, stream>>>(Wq, wtq);
  transpose_w_bf16<<<tgrid, 256, 0, stream>>>(Wk, wtk);
  transpose_w_bf16<<<tgrid, 256, 0, stream>>>(Wv, wtv);
  transpose_w_bf16<<<tgrid, 256, 0, stream>>>(Wo, wto);

  dim3 blk(256);
  dim3 grid(GN / 128, GM / 128);  // (8, 256)
  gemm_bf16_wmma<false, true><<<grid, blk, 0, stream>>>(xb, wtq, nullptr, qb);
  gemm_bf16_wmma<false, true><<<grid, blk, 0, stream>>>(xb, wtk, nullptr, kb);
  gemm_bf16_wmma<false, true><<<grid, blk, 0, stream>>>(xb, wtv, nullptr, vb);

  attn_headmix<<<GM / 4, 128, 0, stream>>>(qb, kb, vb, o2b);

  gemm_bf16_wmma<true, false><<<grid, blk, 0, stream>>>(o2b, wto, bo, (float*)d_out);
}
